// ImprovedTransformerBlock_45028437131477
// MI455X (gfx1250) — matrix operand order repaired and checked by reading, not compiled, not run on hardware
//
#include <hip/hip_runtime.h>

typedef __attribute__((ext_vector_type(16))) __bf16 v16bf;
typedef __attribute__((ext_vector_type(8)))  float  v8f;
typedef __attribute__((ext_vector_type(8)))  unsigned short us8;

union Frag { unsigned u[8]; v16bf v; };

__device__ inline unsigned short f2bf(float f) {
  unsigned u = __builtin_bit_cast(unsigned, f);
  unsigned r = (u + 0x7FFFu + ((u >> 16) & 1u)) >> 16;
  return (unsigned short)r;
}

__device__ inline v8f wmma_bf16(const Frag& a, const Frag& b, v8f c) {
  return __builtin_amdgcn_wmma_f32_16x16x32_bf16(false, a.v, false, b.v,
                                                 (short)0, c, false, false);
}

// LDS byte offset of a __shared__ object (generic -> as(3) -> int)
__device__ inline unsigned lds_off(const void* p) {
  return (unsigned)(unsigned long long)(__attribute__((address_space(3))) const void*)p;
}

// Async global->LDS copy of 16 bytes per lane (tracked by ASYNCcnt)
__device__ inline void async_b128(const void* gsrc, unsigned ldst) {
  asm volatile("global_load_async_to_lds_b128 %0, %1, off"
               :: "v"(ldst), "v"((unsigned long long)gsrc)
               : "memory");
}
__device__ inline void wait_async0() {
  asm volatile("s_wait_asynccnt 0x0" ::: "memory");
}

// ---------------- Tensor Data Mover (TDM) 2D tile load, ISA 08 s8.3/8.4 ----
#if __has_builtin(__builtin_amdgcn_tensor_load_to_lds) && \
    __has_builtin(__builtin_amdgcn_s_wait_tensorcnt)
#define HAVE_TDM 1
typedef __attribute__((ext_vector_type(4))) unsigned int u32x4;
typedef __attribute__((ext_vector_type(8))) int i32x8;
typedef __attribute__((ext_vector_type(4))) int i32x4;

// Load a tile_d1 x tile_d0 (rows x elems) 16-bit tile, row stride stride0
// elems, into LDS at lds_addr, optionally inserting LDS padding:
// pad_amount+1 DWORDs after every 2<<pad_interval DWORDs stored.
__device__ inline void tdm_load_2d_b16(unsigned lds_addr, const void* gaddr,
                                       unsigned tensor_d0, unsigned tensor_d1,
                                       unsigned tile_d0, unsigned tile_d1,
                                       unsigned long long stride0,
                                       unsigned pad_interval, unsigned pad_amount,
                                       int pad_enable) {
  unsigned long long ga = (unsigned long long)gaddr;
  u32x4 g0;
  g0[0] = 1u;                                               // count=1, user mode
  g0[1] = lds_addr;                                         // lds_addr [63:32]
  g0[2] = (unsigned)(ga & 0xFFFFFFFFu);                     // gaddr [95:64]
  g0[3] = (unsigned)((ga >> 32) & 0x01FFFFFFu) | (2u << 30);// gaddr hi + type=2
  unsigned d0 = (1u << 16);                                 // data_size=1 (2B)
  if (pad_enable) d0 |= (1u << 20) | (pad_interval << 22) | (pad_amount << 25);
  unsigned d1 = (tensor_d0 & 0xFFFFu) << 16;                // tensor_dim0 lo
  unsigned d2 = (tensor_d0 >> 16) | ((tensor_d1 & 0xFFFFu) << 16);
  unsigned d3 = (tensor_d1 >> 16) | (tile_d0 << 16);        // tile_dim0
  unsigned d4 = tile_d1;                                    // tile_dim1 (tile2=0)
  unsigned d5 = (unsigned)(stride0 & 0xFFFFFFFFu);          // dim0_stride lo
  unsigned d6 = (unsigned)((stride0 >> 32) & 0xFFFFu);      // dim0_stride hi
  i32x8 g1 = {(int)d0, (int)d1, (int)d2, (int)d3,
              (int)d4, (int)d5, (int)d6, 0};
  i32x4 z4 = {0, 0, 0, 0};
  i32x8 z8 = {0, 0, 0, 0, 0, 0, 0, 0};
  __builtin_amdgcn_tensor_load_to_lds(g0, g1, z4, z4, z8, 0);
}
__device__ inline void wait_tensor0() { __builtin_amdgcn_s_wait_tensorcnt(0); }
#else
#define HAVE_TDM 0
#endif

// ---------------------------------------------------------------- convert
__global__ void k_f32_to_bf16(const float* __restrict__ in,
                              unsigned short* __restrict__ out, long n) {
  long i = (long)blockIdx.x * blockDim.x + threadIdx.x;
  long stride = (long)gridDim.x * blockDim.x;
  for (; i < n; i += stride) out[i] = f2bf(in[i]);
}

// ---------------------------------------------------------------- layernorm -> bf16
__global__ __launch_bounds__(256) void k_layernorm_bf16(
    const float* __restrict__ x, const float* __restrict__ scale,
    unsigned short* __restrict__ out, int D) {
  int row = blockIdx.x;
  const float* xr = x + (long)row * D;
  float s = 0.f, s2 = 0.f;
  for (int i = threadIdx.x; i < D; i += 256) {
    float v = xr[i]; s += v; s2 += v * v;
  }
  for (int m = 1; m < 32; m <<= 1) {
    s += __shfl_xor(s, m, 32);
    s2 += __shfl_xor(s2, m, 32);
  }
  __shared__ float rs[8], rs2[8];
  int w = threadIdx.x >> 5;
  if ((threadIdx.x & 31) == 0) { rs[w] = s; rs2[w] = s2; }
  __syncthreads();
  s = 0.f; s2 = 0.f;
  #pragma unroll
  for (int i = 0; i < 8; i++) { s += rs[i]; s2 += rs2[i]; }
  float inv = 1.0f / (float)D;
  float mean = s * inv;
  float var = s2 * inv - mean * mean;
  float rstd = rsqrtf(var + 1e-6f);
  unsigned short* orow = out + (long)row * D;
  for (int i = threadIdx.x; i < D; i += 256)
    orow[i] = f2bf((xr[i] - mean) * rstd * scale[i]);
}

// ---------------------------------------------------------------- per-head rmsnorm -> bf16
__global__ __launch_bounds__(256) void k_rmsnorm_head(
    const float* __restrict__ x, const float* __restrict__ scale,
    unsigned short* __restrict__ out, float extra, long nvec) {
  long vec = (long)blockIdx.x * 8 + (threadIdx.x >> 5);
  if (vec >= nvec) return;
  int lane = threadIdx.x & 31;
  const float* xr = x + vec * 64;
  float a = xr[lane], b = xr[lane + 32];
  float ss = a * a + b * b;
  for (int m = 1; m < 32; m <<= 1) ss += __shfl_xor(ss, m, 32);
  float r = rsqrtf(ss * (1.0f / 64.0f) + 1e-6f) * extra;
  out[vec * 64 + lane]      = f2bf(a * r * scale[lane]);
  out[vec * 64 + lane + 32] = f2bf(b * r * scale[lane + 32]);
}

// ---------------------------------------------------------------- GEMM (bf16 in, f32 acc)
// Templated epilogue; double-buffered LDS; A tile staged by the TDM (one
// descriptor per K-step, LDS row padding done by the TDM pad engine).
#define BM 128
#define BN 128
#define BK 32
template <bool HB, bool HR, bool GELU, bool OF, bool OBF>
__global__ __launch_bounds__(256) void k_gemm_bf16(
    const unsigned short* __restrict__ A, const unsigned short* __restrict__ B,
    int M, int N, int K,
    const float* __restrict__ bias, const float* __restrict__ res,
    float* __restrict__ outf, unsigned short* __restrict__ outbf) {
  __shared__ unsigned short As[2][BM][BK + 8];   // row stride 80B (64B data + 16B pad)
  __shared__ unsigned short Bs[2][BN][BK + 4];   // transposed, row stride 72B
  int m0 = blockIdx.y * BM;
  int n0 = blockIdx.x * BN;
  int t = threadIdx.x;
  int lane = t & 31, wave = t >> 5;
  int wr = wave >> 1, wc = wave & 1;             // 4x2 wave grid
  int m16 = lane & 15, half = lane >> 4;

  int bk = t >> 3, bns = (t & 7) * 16;           // B stage mapping

  v8f acc[2][4];
  #pragma unroll
  for (int i = 0; i < 2; i++)
    #pragma unroll
    for (int j = 0; j < 4; j++) { v8f z = {}; acc[i][j] = z; }

  us8 bd0, bd1;
  auto stageA = [&](int k0, int buf) {
#if HAVE_TDM
    if (wave == 0)
      tdm_load_2d_b16(lds_off(&As[buf][0][0]), A + (long)m0 * K + k0,
                      (unsigned)K, (unsigned)M, BK, BM,
                      (unsigned long long)K,
                      /*16 DWORDs between pads*/ 3, /*pad 4 DWORDs*/ 3, 1);
#else
    int arow = t >> 1, aseg = (t & 1) * 16;
    const unsigned short* src = A + (long)(m0 + arow) * K + k0 + aseg;
    unsigned dst = lds_off(&As[buf][arow][aseg]);
    async_b128(src, dst);
    async_b128(src + 8, dst + 16);
#endif
  };
  auto waitA = [&]() {
#if HAVE_TDM
    if (wave == 0) wait_tensor0();
#else
    wait_async0();
#endif
  };
  auto loadB = [&](int k0) {
    const unsigned short* src = B + (long)(k0 + bk) * N + n0 + bns;
    bd0 = *(const us8*)src;
    bd1 = *(const us8*)(src + 8);
  };
  auto storeB = [&](int buf) {
    #pragma unroll
    for (int i = 0; i < 8; i++) {
      Bs[buf][bns + i][bk] = bd0[i];
      Bs[buf][bns + 8 + i][bk] = bd1[i];
    }
  };

  stageA(0, 0);
  loadB(0);
  storeB(0);
  waitA();
  __syncthreads();

  int buf = 0;
  for (int k0 = 0; k0 < K; k0 += BK) {
    bool more = (k0 + BK) < K;
    if (more) { stageA(k0 + BK, buf ^ 1); loadB(k0 + BK); }

    Frag af[2], bfr[4];
    #pragma unroll
    for (int mb = 0; mb < 2; mb++) {
      int row = wr * 32 + mb * 16 + m16;
      int kb = half * 8;
      #pragma unroll
      for (int j = 0; j < 4; j++) af[mb].u[j] = *(const unsigned*)&As[buf][row][kb + 2 * j];
      #pragma unroll
      for (int j = 0; j < 4; j++) af[mb].u[4 + j] = *(const unsigned*)&As[buf][row][kb + 16 + 2 * j];
    }
    #pragma unroll
    for (int nb = 0; nb < 4; nb++) {
      int col = wc * 64 + nb * 16 + m16;
      int kb = half * 8;
      #pragma unroll
      for (int j = 0; j < 8; j++) bfr[nb].u[j] = *(const unsigned*)&Bs[buf][col][kb + (j >> 2) * 16 + 2 * (j & 3)];
    }
    #pragma unroll
    for (int mb = 0; mb < 2; mb++)
      #pragma unroll
      for (int nb = 0; nb < 4; nb++)
        acc[mb][nb] = wmma_bf16(af[mb], bfr[nb], acc[mb][nb]);

    if (more) storeB(buf ^ 1);
    waitA();
    __syncthreads();
    buf ^= 1;
  }

  #pragma unroll
  for (int mb = 0; mb < 2; mb++) {
    #pragma unroll
    for (int nb = 0; nb < 4; nb++) {
      int col = n0 + wc * 64 + nb * 16 + m16;
      float bv = 0.0f;
      if constexpr (HB) bv = bias[col];
      #pragma unroll
      for (int r = 0; r < 8; r++) {
        int row = m0 + wr * 32 + mb * 16 + r + 8 * half;
        long idx = (long)row * N + col;
        float v = acc[mb][nb][r] + bv;
        if constexpr (HR) v += res[idx];
        if constexpr (GELU) {
          float x3 = v * v * v;
          float tt = tanhf(0.7978845608028654f * (v + 0.044715f * x3));
          v = 0.5f * v * (1.0f + tt);
        }
        if constexpr (OF) outf[idx] = v;
        if constexpr (OBF) outbf[idx] = f2bf(v);
      }
    }
  }
}

// ---------------------------------------------------------------- flash attention
// Q,K,V,O: [B, N, H, 64] bf16. Block = 8 waves sharing (b,h); each wave owns a
// 16-row q block (128 q rows per block). K staged via async-to-LDS,
// V staged transposed, kv tiles of 32, online softmax in registers.
__global__ __launch_bounds__(256) void k_flash_attn(
    const unsigned short* __restrict__ Q, const unsigned short* __restrict__ Km,
    const unsigned short* __restrict__ Vm, unsigned short* __restrict__ O,
    int Nq, int Nk, int H) {
  __shared__ unsigned short Ks[32][72];        // [kv][e], row stride 144B
  __shared__ unsigned short VT[64][36];        // [e][kv], row stride 72B
  __shared__ unsigned short Pl[8][16][BK + 4]; // per-wave P staging
  int t = threadIdx.x;
  int wave = t >> 5, lane = t & 31;
  int qb = blockIdx.x % (Nq / 128);
  int h = (blockIdx.x / (Nq / 128)) % H;
  int b = blockIdx.x / ((Nq / 128) * H);
  const int HD = 64;
  int rowstride = H * HD;
  long qbase = (((long)b * Nq + (long)qb * 128 + wave * 16) * H + h) * HD;
  long kbase = ((long)b * Nk * H + h) * HD;
  int m = lane & 15, half = lane >> 4;
  const float LOG2E = 1.4426950408889634f;

  // Q fragments (A layout), loaded once from global
  Frag qf[2];
  const unsigned short* qrow = Q + qbase + (long)m * rowstride;
  #pragma unroll
  for (int es = 0; es < 2; es++) {
    int e0 = es * 32 + half * 8;
    #pragma unroll
    for (int j = 0; j < 4; j++) qf[es].u[j] = *(const unsigned*)(qrow + e0 + 2 * j);
    #pragma unroll
    for (int j = 0; j < 4; j++) qf[es].u[4 + j] = *(const unsigned*)(qrow + e0 + 16 + 2 * j);
  }

  v8f acc[4];
  float mi[8], li[8];
  #pragma unroll
  for (int nb = 0; nb < 4; nb++) { v8f z = {}; acc[nb] = z; }
  #pragma unroll
  for (int r = 0; r < 8; r++) { mi[r] = -3.0e38f; li[r] = 0.f; }

  int krow = t >> 3, kseg = (t & 7) * 8;       // K stage: 256 x 16B segments
  int vrow = t >> 3, vseg = (t & 7) * 8;       // V stage: 8 elems/thread

  for (int kt = 0; kt < Nk; kt += 32) {
    __syncthreads();   // previous tile fully consumed before restaging
    async_b128(Km + kbase + (long)(kt + krow) * rowstride + kseg,
               lds_off(&Ks[krow][kseg]));
    {
      const unsigned short* src = Vm + kbase + (long)(kt + vrow) * rowstride + vseg;
      us8 d0 = *(const us8*)src;
      #pragma unroll
      for (int i = 0; i < 8; i++) VT[vseg + i][vrow] = d0[i];
    }
    wait_async0();
    __syncthreads();

    // S = Q * K^T for two 16-col kv blocks
    v8f s[2];
    #pragma unroll
    for (int cb = 0; cb < 2; cb++) {
      v8f sv = {};
      int n = cb * 16 + m;
      #pragma unroll
      for (int es = 0; es < 2; es++) {
        Frag kf;
        int e0 = es * 32 + half * 8;
        #pragma unroll
        for (int j = 0; j < 8; j++) kf.u[j] = *(const unsigned*)&Ks[n][e0 + (j >> 2) * 16 + 2 * (j & 3)];
        sv = wmma_bf16(qf[es], kf, sv);
      }
      s[cb] = sv;
    }
    // online softmax (rows per-vgpr; 16 cols across lanes of each half)
    float alpha[8];
    #pragma unroll
    for (int r = 0; r < 8; r++) {
      float mx = fmaxf(s[0][r], s[1][r]);
      mx = fmaxf(mx, __shfl_xor(mx, 1, 32));
      mx = fmaxf(mx, __shfl_xor(mx, 2, 32));
      mx = fmaxf(mx, __shfl_xor(mx, 4, 32));
      mx = fmaxf(mx, __shfl_xor(mx, 8, 32));
      float mnew = fmaxf(mi[r], mx);
      float a = exp2f((mi[r] - mnew) * LOG2E);
      float p0 = exp2f((s[0][r] - mnew) * LOG2E);
      float p1 = exp2f((s[1][r] - mnew) * LOG2E);
      float rsum = p0 + p1;
      rsum += __shfl_xor(rsum, 1, 32);
      rsum += __shfl_xor(rsum, 2, 32);
      rsum += __shfl_xor(rsum, 4, 32);
      rsum += __shfl_xor(rsum, 8, 32);
      li[r] = li[r] * a + rsum;
      mi[r] = mnew;
      alpha[r] = a;
      s[0][r] = p0;
      s[1][r] = p1;
    }
    #pragma unroll
    for (int nb = 0; nb < 4; nb++)
      #pragma unroll
      for (int r = 0; r < 8; r++)
        acc[nb][r] *= alpha[r];
    // stage P (C layout -> per-wave LDS slice) then regather in A layout
    #pragma unroll
    for (int cb = 0; cb < 2; cb++)
      #pragma unroll
      for (int r = 0; r < 8; r++)
        Pl[wave][r + 8 * half][cb * 16 + m] = f2bf(s[cb][r]);
    asm volatile("s_wait_dscnt 0x0" ::: "memory");
    __builtin_amdgcn_wave_barrier();
    Frag pf;
    {
      int k0 = half * 8;
      #pragma unroll
      for (int j = 0; j < 4; j++) pf.u[j] = *(const unsigned*)&Pl[wave][m][k0 + 2 * j];
      #pragma unroll
      for (int j = 0; j < 4; j++) pf.u[4 + j] = *(const unsigned*)&Pl[wave][m][k0 + 16 + 2 * j];
    }
    asm volatile("s_wait_dscnt 0x0" ::: "memory");
    __builtin_amdgcn_wave_barrier();
    // acc += P * V (V fragments from transposed LDS: kv 8h..8h+7 and 16+8h..16+8h+7 in lane half h)
    #pragma unroll
    for (int nb = 0; nb < 4; nb++) {
      Frag vf;
      int e = nb * 16 + m;
      int kb = half * 8;
      #pragma unroll
      for (int j = 0; j < 8; j++) vf.u[j] = *(const unsigned*)&VT[e][kb + (j >> 2) * 16 + 2 * (j & 3)];
      acc[nb] = wmma_bf16(pf, vf, acc[nb]);
    }
  }
  #pragma unroll
  for (int nb = 0; nb < 4; nb++) {
    #pragma unroll
    for (int r = 0; r < 8; r++) {
      int row = r + 8 * half;
      float v = acc[nb][r] / li[r];
      O[qbase + (long)row * rowstride + nb * 16 + m] = f2bf(v);
    }
  }
}

// ---------------------------------------------------------------- host orchestration
static inline size_t alignup(size_t x) { return (x + 255) & ~(size_t)255; }

extern "C" void kernel_launch(void* const* d_in, const int* in_sizes, int n_in,
                              void* d_out, int out_size, void* d_ws, size_t ws_size,
                              hipStream_t stream) {
  (void)in_sizes; (void)n_in; (void)out_size; (void)ws_size;
  const int B = 2, N = 2048, D = 1024, H = 16, Mhid = 4096;
  const int R = B * N;

  const float* queries   = (const float*)d_in[0];
  const float* inputs_kv = (const float*)d_in[1];
  const float* norm_q_s  = (const float*)d_in[2];
  const float* sa_wq = (const float*)d_in[3];
  const float* sa_wk = (const float*)d_in[4];
  const float* sa_wv = (const float*)d_in[5];
  const float* sa_qn = (const float*)d_in[6];
  const float* sa_kn = (const float*)d_in[7];
  const float* sa_wo = (const float*)d_in[8];
  const float* sa_bo = (const float*)d_in[9];
  const float* ca_wq = (const float*)d_in[10];
  const float* ca_wk = (const float*)d_in[11];
  const float* ca_wv = (const float*)d_in[12];
  const float* ca_qn = (const float*)d_in[13];
  const float* ca_kn = (const float*)d_in[14];
  const float* ca_wo = (const float*)d_in[15];
  const float* ca_bo = (const float*)d_in[16];
  const float* norm_a_s  = (const float*)d_in[17];
  const float* w_in  = (const float*)d_in[18];
  const float* b_in  = (const float*)d_in[19];
  const float* w_out = (const float*)d_in[20];
  const float* b_out = (const float*)d_in[21];
  float* out = (float*)d_out;

  char* ws = (char*)d_ws;
  size_t off = 0;
  auto take = [&](size_t bytes) { char* p = ws + off; off += alignup(bytes); return p; };
  unsigned short* nq_bf  = (unsigned short*)take((size_t)R * D * 2);
  unsigned short* kv_bf  = (unsigned short*)take((size_t)R * D * 2);
  unsigned short* wq_bf  = (unsigned short*)take((size_t)D * D * 2);
  unsigned short* wk_bf  = (unsigned short*)take((size_t)D * D * 2);
  unsigned short* wv_bf  = (unsigned short*)take((size_t)D * D * 2);
  unsigned short* wo_bf  = (unsigned short*)take((size_t)D * D * 2);
  unsigned short* cwq_bf = (unsigned short*)take((size_t)D * D * 2);
  unsigned short* cwk_bf = (unsigned short*)take((size_t)D * D * 2);
  unsigned short* cwv_bf = (unsigned short*)take((size_t)D * D * 2);
  unsigned short* cwo_bf = (unsigned short*)take((size_t)D * D * 2);
  unsigned short* win_bf  = (unsigned short*)take((size_t)D * Mhid * 2);
  unsigned short* wout_bf = (unsigned short*)take((size_t)Mhid * D * 2);
  float* qf = (float*)take((size_t)R * D * 4);
  float* kf = (float*)take((size_t)R * D * 4);
  unsigned short* q_bf = (unsigned short*)take((size_t)R * D * 2);
  unsigned short* k_bf = (unsigned short*)take((size_t)R * D * 2);
  unsigned short* v_bf = (unsigned short*)take((size_t)R * D * 2);
  unsigned short* x_bf = (unsigned short*)take((size_t)R * D * 2);
  float* attn_tmp = (float*)take((size_t)R * D * 4);
  float* attn_out = (float*)take((size_t)R * D * 4);
  unsigned short* ln2_bf = (unsigned short*)take((size_t)R * D * 2);
  unsigned short* h_bf   = (unsigned short*)take((size_t)R * Mhid * 2);

  auto conv = [&](const float* src, unsigned short* dst, long n) {
    int blocks = (int)((n + 1023) / 1024);
    if (blocks > 2048) blocks = 2048;
    k_f32_to_bf16<<<blocks, 256, 0, stream>>>(src, dst, n);
  };
  // mode 0: f32 out; 1: bf16 out; 2: bias+res+f32; 3: bias+gelu+bf16
  auto gemm = [&](int mode, const unsigned short* A, const unsigned short* Bm,
                  int M_, int N_, int K_, const float* bias, const float* res,
                  float* outf, unsigned short* outbf) {
    dim3 grid(N_ / BN, M_ / BM);
    switch (mode) {
      case 0: k_gemm_bf16<false, false, false, true, false>
                  <<<grid, 256, 0, stream>>>(A, Bm, M_, N_, K_, bias, res, outf, outbf); break;
      case 1: k_gemm_bf16<false, false, false, false, true>
                  <<<grid, 256, 0, stream>>>(A, Bm, M_, N_, K_, bias, res, outf, outbf); break;
      case 2: k_gemm_bf16<true, true, false, true, false>
                  <<<grid, 256, 0, stream>>>(A, Bm, M_, N_, K_, bias, res, outf, outbf); break;
      case 3: k_gemm_bf16<true, false, true, false, true>
                  <<<grid, 256, 0, stream>>>(A, Bm, M_, N_, K_, bias, res, outf, outbf); break;
    }
  };

  conv(inputs_kv, kv_bf, (long)R * D);
  conv(sa_wq, wq_bf, (long)D * D);  conv(sa_wk, wk_bf, (long)D * D);
  conv(sa_wv, wv_bf, (long)D * D);  conv(sa_wo, wo_bf, (long)D * D);
  conv(ca_wq, cwq_bf, (long)D * D); conv(ca_wk, cwk_bf, (long)D * D);
  conv(ca_wv, cwv_bf, (long)D * D); conv(ca_wo, cwo_bf, (long)D * D);
  conv(w_in, win_bf, (long)D * Mhid); conv(w_out, wout_bf, (long)Mhid * D);

  k_layernorm_bf16<<<R, 256, 0, stream>>>(queries, norm_q_s, nq_bf, D);

  const long nvec = (long)R * H;
  const int rms_blocks = (int)(nvec / 8);
  const int fa_blocks = B * H * (N / 128);
  const float qscale = 0.125f;   // 1/sqrt(64)

  // ---- self attention
  gemm(0, nq_bf, wq_bf, R, D, D, nullptr, nullptr, qf, nullptr);
  gemm(0, nq_bf, wk_bf, R, D, D, nullptr, nullptr, kf, nullptr);
  gemm(1, nq_bf, wv_bf, R, D, D, nullptr, nullptr, nullptr, v_bf);
  k_rmsnorm_head<<<rms_blocks, 256, 0, stream>>>(qf, sa_qn, q_bf, qscale, nvec);
  k_rmsnorm_head<<<rms_blocks, 256, 0, stream>>>(kf, sa_kn, k_bf, 1.0f, nvec);
  k_flash_attn<<<fa_blocks, 256, 0, stream>>>(q_bf, k_bf, v_bf, x_bf, N, N, H);
  gemm(2, x_bf, wo_bf, R, D, D, sa_bo, queries, attn_tmp, nullptr);

  // ---- cross attention
  gemm(0, nq_bf, cwq_bf, R, D, D, nullptr, nullptr, qf, nullptr);
  gemm(0, kv_bf, cwk_bf, R, D, D, nullptr, nullptr, kf, nullptr);
  gemm(1, kv_bf, cwv_bf, R, D, D, nullptr, nullptr, nullptr, v_bf);
  k_rmsnorm_head<<<rms_blocks, 256, 0, stream>>>(qf, ca_qn, q_bf, qscale, nvec);
  k_rmsnorm_head<<<rms_blocks, 256, 0, stream>>>(kf, ca_kn, k_bf, 1.0f, nvec);
  k_flash_attn<<<fa_blocks, 256, 0, stream>>>(q_bf, k_bf, v_bf, x_bf, N, N, H);
  gemm(2, x_bf, cwo_bf, R, D, D, ca_bo, attn_tmp, attn_out, nullptr);

  // ---- MLP
  k_layernorm_bf16<<<R, 256, 0, stream>>>(attn_out, norm_a_s, ln2_bf, D);
  gemm(3, ln2_bf, win_bf, R, Mhid, D, b_in, nullptr, nullptr, h_bf);
  gemm(2, h_bf, wout_bf, R, D, Mhid, b_out, attn_out, out, nullptr);
}
